// LSTMModel_71536975282811
// MI455X (gfx1250) — matrix operand order repaired and checked by reading, not compiled, not run on hardware
//
#include <hip/hip_runtime.h>
#include <hip/hip_bf16.h>

// ---------------- types ----------------
typedef __attribute__((ext_vector_type(16))) __bf16 v16bf;
typedef __attribute__((ext_vector_type(8)))  float  v8f;
typedef __attribute__((ext_vector_type(4)))  unsigned int v4u;

// ---------------- problem dims ----------------
constexpr int BATCH = 64;
constexpr int SEQ   = 256;
constexpr int EMBD  = 512;
constexpr int HIDD  = 1024;
constexpr int OUTD  = 512;
constexpr int KTOT  = HIDD + EMBD;   // 1536: [h | x_t]
constexpr int NBLK  = 32;            // persistent grid: 32 blocks x 8 wave32 = 256 waves
constexpr int NTHR  = 256;
constexpr int APITCH = KTOT + 8;     // LDS A-panel row pitch (bf16): 3088B = 772 DW == 4 mod 64 banks

// ---------------- workspace layout (bytes) ----------------
constexpr size_t WCOMB_OFF = 0;                                   // [4][HIDD][KTOT] bf16 (W_h;W_x)^T per gate-column
constexpr size_t WCOMB_SZ  = (size_t)4 * HIDD * KTOT * 2;         // 12 MB
constexpr size_t WHYT_OFF  = WCOMB_OFF + WCOMB_SZ;                // [OUTD][HIDD] bf16 = W_hy^T
constexpr size_t WHYT_SZ   = (size_t)OUTD * HIDD * 2;             // 1 MB
constexpr size_t XBF_OFF   = WHYT_OFF + WHYT_SZ;                  // [BATCH][SEQ][EMBD] bf16 gathered embeddings
constexpr size_t XBF_SZ    = (size_t)BATCH * SEQ * EMBD * 2;      // 16 MB
constexpr size_t HBUF_OFF  = XBF_OFF + XBF_SZ;                    // [BATCH][HIDD] bf16 hidden state
constexpr size_t HBUF_SZ   = (size_t)BATCH * HIDD * 2;            // 128 KB
constexpr size_t BAR_OFF   = HBUF_OFF + HBUF_SZ;                  // 2 x u32 barrier state

// ---------------- helpers ----------------
__device__ __forceinline__ unsigned short f2bf(float f) {
  unsigned int u = __float_as_uint(f);
  u += 0x7FFFu + ((u >> 16) & 1u);     // round-to-nearest-even
  return (unsigned short)(u >> 16);
}

__device__ __forceinline__ v16bf ld32(const unsigned short* p0, const unsigned short* p1) {
  union { v4u u[2]; v16bf v; } r;
  r.u[0] = *(const v4u*)p0;   // global_load_b128 / ds_load_b128
  r.u[1] = *(const v4u*)p1;
  return r.v;
}

__device__ __forceinline__ float sigmoid_(float x) { return 1.0f / (1.0f + __expf(-x)); }
__device__ __forceinline__ float tanh_(float x)    { float e = __expf(2.0f * x); return 1.0f - 2.0f / (e + 1.0f); }

// async copy 16B global -> LDS (gfx1250 ASYNCcnt path)
__device__ __forceinline__ void async_g2l_b128(unsigned ldsoff, const void* gptr) {
  unsigned long long g = (unsigned long long)(uintptr_t)gptr;
  asm volatile("global_load_async_to_lds_b128 %0, %1, off" :: "v"(ldsoff), "v"(g) : "memory");
}
__device__ __forceinline__ void wait_asynccnt0() {
  asm volatile("s_wait_asynccnt 0x0" ::: "memory");
}

__device__ __forceinline__ void grid_sync(unsigned int* bar) {
  __threadfence();
  __syncthreads();
  if (threadIdx.x == 0) {
    unsigned int gen = __hip_atomic_load(&bar[1], __ATOMIC_RELAXED, __HIP_MEMORY_SCOPE_AGENT);
    unsigned int arrived =
        __hip_atomic_fetch_add(&bar[0], 1u, __ATOMIC_ACQ_REL, __HIP_MEMORY_SCOPE_AGENT) + 1u;
    if (arrived == (unsigned)NBLK) {
      __hip_atomic_store(&bar[0], 0u, __ATOMIC_RELAXED, __HIP_MEMORY_SCOPE_AGENT);
      __hip_atomic_store(&bar[1], gen + 1u, __ATOMIC_RELEASE, __HIP_MEMORY_SCOPE_AGENT);
    } else {
      while (__hip_atomic_load(&bar[1], __ATOMIC_ACQUIRE, __HIP_MEMORY_SCOPE_AGENT) == gen) {
        __builtin_amdgcn_s_sleep(1);
      }
    }
  }
  __syncthreads();
  __threadfence();
}

// ---------------- prep kernels ----------------
// Wcomb[g][j][k] = (k<HIDD) ? W_h{g}[k][j] : W_x{g}[k-HIDD][j]   (bf16, gate order i,f,o,c)
__global__ void k_build_wcomb(const float* __restrict__ whi, const float* __restrict__ whf,
                              const float* __restrict__ who, const float* __restrict__ whc,
                              const float* __restrict__ wxi, const float* __restrict__ wxf,
                              const float* __restrict__ wxo, const float* __restrict__ wxc,
                              unsigned short* __restrict__ dst) {
  size_t idx = (size_t)blockIdx.x * NTHR + threadIdx.x;
  int g   = (int)(idx / ((size_t)HIDD * KTOT));
  size_t rem = idx % ((size_t)HIDD * KTOT);
  int j = (int)(rem / KTOT);
  int k = (int)(rem % KTOT);
  const float* wh = (g == 0) ? whi : (g == 1) ? whf : (g == 2) ? who : whc;
  const float* wx = (g == 0) ? wxi : (g == 1) ? wxf : (g == 2) ? wxo : wxc;
  float v = (k < HIDD) ? wh[(size_t)k * HIDD + j] : wx[(size_t)(k - HIDD) * HIDD + j];
  dst[idx] = f2bf(v);
}

__global__ void k_build_whyT(const float* __restrict__ why, unsigned short* __restrict__ dst) {
  size_t idx = (size_t)blockIdx.x * NTHR + threadIdx.x;
  int n = (int)(idx / HIDD);
  int k = (int)(idx % HIDD);
  dst[idx] = f2bf(why[(size_t)k * OUTD + n]);
}

__global__ void k_embed(const int* __restrict__ tokens, const float* __restrict__ emb,
                        unsigned short* __restrict__ dst) {
  size_t idx = (size_t)blockIdx.x * NTHR + threadIdx.x;
  int e  = (int)(idx & (EMBD - 1));
  size_t bs = idx >> 9;                 // b*SEQ + t
  int tok = tokens[bs];
  dst[idx] = f2bf(emb[(size_t)tok * EMBD + e]);
}

__global__ void k_init(unsigned short* __restrict__ hbuf, unsigned int* __restrict__ bar) {
  size_t idx = (size_t)blockIdx.x * NTHR + threadIdx.x;
  hbuf[idx] = 0;
  if (idx == 0) { bar[0] = 0u; bar[1] = 0u; }
}

// ---------------- persistent fused LSTM ----------------
__global__ __launch_bounds__(NTHR) void lstm_persistent(
    const unsigned short* __restrict__ wcomb,
    const unsigned short* __restrict__ whyT,
    const unsigned short* __restrict__ xbf,
    unsigned short* __restrict__ hbuf,
    unsigned int* __restrict__ bar,
    const float* __restrict__ b_i, const float* __restrict__ b_f,
    const float* __restrict__ b_o, const float* __restrict__ b_c,
    const float* __restrict__ b_hy, float* __restrict__ out) {
  // Block-shared A panel: 16 rows (batch) x [h(1024) | x_t(512)], bf16, padded pitch.
  __shared__ unsigned short sA[16 * APITCH];

  const int tid  = threadIdx.x;
  const int lane = tid & 31;
  const int wid  = tid >> 5;
  const int gw   = blockIdx.x * (NTHR / 32) + wid;   // 0..255: one 16x16 (M x H) tile per wave
  const int lrow = lane & 15;
  const int lhi  = lane >> 4;

  const int m  = gw >> 6;                 // batch tile 0..3 (constant within a block)
  const int jt = gw & 63;                 // hidden tile 0..63
  const int mbase = m * 16;
  const int colH  = jt * 16 + lrow;       // this lane's N (hidden) column

  const float bi = b_i[colH], bff = b_f[colH], bo = b_o[colH], bc = b_c[colH];

  // per-gate B-operand column bases: [g][colH][k], contiguous in k
  const unsigned short* wg0 = wcomb + ((size_t)(0 * HIDD + colH)) * KTOT;
  const unsigned short* wg1 = wcomb + ((size_t)(1 * HIDD + colH)) * KTOT;
  const unsigned short* wg2 = wcomb + ((size_t)(2 * HIDD + colH)) * KTOT;
  const unsigned short* wg3 = wcomb + ((size_t)(3 * HIDD + colH)) * KTOT;

  const unsigned sAbase = (unsigned)(uintptr_t)(void*)&sA[0];   // LDS byte offset of panel

  v8f cst = {};                           // cell state tile, resident in VGPRs for all 256 steps

  for (int t = 0; t < SEQ; ++t) {
    // ---- async-stage A panel into LDS (shared by all 8 waves of this block) ----
    // h part: 16 rows x 1024 cols = 2048 x 16B chunks
    #pragma unroll
    for (int i = 0; i < 8; ++i) {
      int c   = tid + NTHR * i;                 // 0..2047
      int row = c >> 7;                         // 128 chunks per row
      int col = (c & 127) * 8;
      unsigned ldsoff = sAbase + (unsigned)(row * APITCH + col) * 2u;
      async_g2l_b128(ldsoff, hbuf + (size_t)(mbase + row) * HIDD + col);
    }
    // x_t part: 16 rows x 512 cols = 1024 x 16B chunks
    #pragma unroll
    for (int i = 0; i < 4; ++i) {
      int c   = tid + NTHR * i;                 // 0..1023
      int row = c >> 6;                         // 64 chunks per row
      int col = (c & 63) * 8;
      unsigned ldsoff = sAbase + (unsigned)(row * APITCH + HIDD + col) * 2u;
      async_g2l_b128(ldsoff, xbf + ((size_t)(mbase + row) * SEQ + t) * EMBD + col);
    }
    wait_asynccnt0();
    __syncthreads();

    v8f ai = {}, af = {}, ao = {}, ac = {};
    const unsigned short* aRow = &sA[lrow * APITCH];   // this lane's A row in LDS

    #pragma unroll 4
    for (int kt = 0; kt < KTOT / 32; ++kt) {
      const int kb = kt * 32;
      const unsigned short* ap = aRow + kb + lhi * 16;
      v16bf a = ld32(ap, ap + 8);                      // ds_load_b128 x2
      const int kob = kb + lhi * 16;
      __builtin_prefetch(wg0 + kob + 128, 0, 3);       // global_prefetch_b8
      v16bf w0 = ld32(wg0 + kob, wg0 + kob + 8);
      ai = __builtin_amdgcn_wmma_f32_16x16x32_bf16(false, a, false, w0, (short)0, ai, false, false);
      v16bf w1 = ld32(wg1 + kob, wg1 + kob + 8);
      af = __builtin_amdgcn_wmma_f32_16x16x32_bf16(false, a, false, w1, (short)0, af, false, false);
      v16bf w2 = ld32(wg2 + kob, wg2 + kob + 8);
      ao = __builtin_amdgcn_wmma_f32_16x16x32_bf16(false, a, false, w2, (short)0, ao, false, false);
      v16bf w3 = ld32(wg3 + kob, wg3 + kob + 8);
      ac = __builtin_amdgcn_wmma_f32_16x16x32_bf16(false, a, false, w3, (short)0, ac, false, false);
    }

    // gate nonlinearities, cell/h update, publish h (bf16) for next step
    #pragma unroll
    for (int r = 0; r < 8; ++r) {
      float gi = sigmoid_(ai[r] + bi);
      float gf = sigmoid_(af[r] + bff);
      float go = sigmoid_(ao[r] + bo);
      float gc = tanh_(ac[r] + bc);
      float cv = gf * cst[r] + gi * gc;
      cst[r] = cv;
      float hv = go * tanh_(cv);
      const int row = mbase + lhi * 8 + r;           // C/D layout: VGPR r -> M = r + 8*lhi
      hbuf[(size_t)row * HIDD + colH] = f2bf(hv);
    }

    grid_sync(bar);   // also serializes LDS reuse across steps (syncthreads inside)
  }

  // ---- final projection: out = h @ W_hy + b_hy  (4 x 32 tiles -> first 128 waves) ----
  if (gw < 128) {
    const int mo = gw >> 5;                 // 0..3
    const int nt = gw & 31;                 // 0..31
    const int colO = nt * 16 + lrow;
    const unsigned short* wy  = whyT + (size_t)colO * HIDD;
    const unsigned short* hAo = hbuf + (size_t)(mo * 16 + lrow) * HIDD;
    v8f acc = {};
    #pragma unroll 4
    for (int kt = 0; kt < HIDD / 32; ++kt) {
      const int kb = kt * 32;
      const unsigned short* ap = hAo + kb + lhi * 16;
      v16bf a = ld32(ap, ap + 8);
      const int kob = kb + lhi * 16;
      v16bf w = ld32(wy + kob, wy + kob + 8);
      acc = __builtin_amdgcn_wmma_f32_16x16x32_bf16(false, a, false, w, (short)0, acc, false, false);
    }
    const float bhy = b_hy[colO];
    #pragma unroll
    for (int r = 0; r < 8; ++r) {
      const int row = mo * 16 + lhi * 8 + r;
      out[(size_t)row * OUTD + colO] = acc[r] + bhy;
    }
  }
}

// ---------------- launch ----------------
extern "C" void kernel_launch(void* const* d_in, const int* in_sizes, int n_in,
                              void* d_out, int out_size, void* d_ws, size_t ws_size,
                              hipStream_t stream) {
  (void)in_sizes; (void)n_in; (void)out_size; (void)ws_size;

  const int*   tokens = (const int*)  d_in[0];
  const float* emb    = (const float*)d_in[1];
  const float* wxi    = (const float*)d_in[2];
  const float* wxf    = (const float*)d_in[3];
  const float* wxo    = (const float*)d_in[4];
  const float* wxc    = (const float*)d_in[5];
  const float* whi    = (const float*)d_in[6];
  const float* whf    = (const float*)d_in[7];
  const float* who    = (const float*)d_in[8];
  const float* whc    = (const float*)d_in[9];
  const float* why    = (const float*)d_in[10];
  const float* b_i    = (const float*)d_in[11];
  const float* b_f    = (const float*)d_in[12];
  const float* b_o    = (const float*)d_in[13];
  const float* b_c    = (const float*)d_in[14];
  const float* b_hy   = (const float*)d_in[15];
  float* out = (float*)d_out;

  char* ws = (char*)d_ws;
  unsigned short* wcomb = (unsigned short*)(ws + WCOMB_OFF);
  unsigned short* whyT  = (unsigned short*)(ws + WHYT_OFF);
  unsigned short* xbf   = (unsigned short*)(ws + XBF_OFF);
  unsigned short* hbuf  = (unsigned short*)(ws + HBUF_OFF);
  unsigned int*   bar   = (unsigned int*)  (ws + BAR_OFF);

  dim3 blk(NTHR);
  k_build_wcomb<<<(4 * HIDD * KTOT) / NTHR, blk, 0, stream>>>(whi, whf, who, whc,
                                                             wxi, wxf, wxo, wxc, wcomb);
  k_build_whyT<<<(OUTD * HIDD) / NTHR, blk, 0, stream>>>(why, whyT);
  k_embed<<<(BATCH * SEQ * EMBD) / NTHR, blk, 0, stream>>>(tokens, emb, xbf);
  k_init<<<(BATCH * HIDD) / NTHR, blk, 0, stream>>>(hbuf, bar);
  lstm_persistent<<<NBLK, blk, 0, stream>>>(wcomb, whyT, xbf, hbuf, bar,
                                            b_i, b_f, b_o, b_c, b_hy, out);
}
